// Framework_12790412607649
// MI455X (gfx1250) — matrix operand order repaired and checked by reading, not compiled, not run on hardware
//
#include <hip/hip_runtime.h>

#define BB 1024
#define NN 131072
#define EE 131072
#define DD 512
#define KK 8
#define HH 256
#define TT 12

#define ROW_PAD 80   // bytes per staged 64B B-row: 16B-aligned, bank-conflict-free stride

typedef __attribute__((ext_vector_type(16))) __bf16 v16bf;
typedef __attribute__((ext_vector_type(8)))  __bf16 v8bf;
typedef __attribute__((ext_vector_type(8)))  float  v8f;

#if __has_builtin(__builtin_amdgcn_global_load_async_to_lds_b128)
#define HAS_ASYNC_LDS 1
#else
#define HAS_ASYNC_LDS 0
#endif

template <int N>
__device__ __forceinline__ void wait_async_le() {
#if HAS_ASYNC_LDS
#if __has_builtin(__builtin_amdgcn_s_wait_asynccnt)
    __builtin_amdgcn_s_wait_asynccnt(N);
#else
    asm volatile("s_wait_asynccnt %0" ::"n"(N) : "memory");
#endif
#endif
}

// Stage one 64-byte row (global bf16 row slice) into LDS at dst + t*ROW_PAD.
__device__ __forceinline__ void stage_b(const __bf16* __restrict__ W1t, char* dst,
                                        int kb, int t) {
    const char* g = (const char*)(W1t + (size_t)t * DD + kb);
    char* l = dst + t * ROW_PAD;
#if HAS_ASYNC_LDS
    typedef __attribute__((ext_vector_type(4))) int v4i;
    typedef __attribute__((address_space(1))) v4i gv4;
    typedef __attribute__((address_space(3))) v4i lv4;
    gv4* gp = (gv4*)(unsigned long long)(const void*)g;
    lv4* lp = (lv4*)(unsigned)(unsigned long long)(void*)l;
    __builtin_amdgcn_global_load_async_to_lds_b128(gp + 0, lp + 0, 0, 0);
    __builtin_amdgcn_global_load_async_to_lds_b128(gp + 1, lp + 1, 0, 0);
    __builtin_amdgcn_global_load_async_to_lds_b128(gp + 2, lp + 2, 0, 0);
    __builtin_amdgcn_global_load_async_to_lds_b128(gp + 3, lp + 3, 0, 0);
#else
    const float4* gf = (const float4*)g;
    float4* lf = (float4*)l;
    lf[0] = gf[0]; lf[1] = gf[1]; lf[2] = gf[2]; lf[3] = gf[3];
#endif
}

// ---------------- prep: f32 src[R][C] -> bf16 dst[C][R] (transpose + convert) ----------------
__global__ void k_transpose_bf16(const float* __restrict__ src, __bf16* __restrict__ dst,
                                 int R, int C) {
    int i = blockIdx.x * blockDim.x + threadIdx.x;
    if (i >= R * C) return;
    int c = i / R, r = i % R;
    dst[(size_t)c * R + r] = (__bf16)src[(size_t)r * C + c];
}

// ---------------- segment bounds over sorted batch_idx ----------------
__global__ void k_seg_init(int* __restrict__ segs, int* __restrict__ sege) {
    int b = blockIdx.x * blockDim.x + threadIdx.x;
    if (b < BB) { segs[b] = 0x7fffffff; sege[b] = 0; }
}

__global__ void k_seg_bounds(const int* __restrict__ batch_idx,
                             int* __restrict__ segs, int* __restrict__ sege) {
    int e = blockIdx.x * blockDim.x + threadIdx.x;
    if (e >= EE) return;
    int b = batch_idx[e];
    atomicMin(&segs[b], e);
    atomicMax(&sege[b], e + 1);
}

// ---------------- fused gather + attn scorer: s[e] = tanh(x@W1+b1)@W2 + b2 ----------------
// One wave per 16-row tile of gathered x. A fragments from global f32 gather (cvt to bf16
// on the fly). B slices (all 256 cols x 32 K) double-buffered in LDS via async-to-LDS DMA.
__global__ __launch_bounds__(256)
void k_score(const float* __restrict__ sub_feat, const int* __restrict__ feat_idx,
             const __bf16* __restrict__ W1t, const float* __restrict__ b1,
             const float* __restrict__ W2, const float* __restrict__ b2,
             float* __restrict__ s_out) {
    __shared__ __align__(16) char bstage[2][256 * ROW_PAD];   // 40 KB double buffer

    const int t    = threadIdx.x;
    const int lane = threadIdx.x & 31;
    const int wid  = threadIdx.x >> 5;
    const int hf   = lane >> 4;        // which 16-lane half
    const int m    = lane & 15;        // row (A) / column (B,C) within tile
    const int tile = blockIdx.x * 8 + wid;
    const int r0   = tile * 16;
    const int row  = feat_idx[r0 + m];
    const float* xr = sub_feat + (size_t)row * DD;

    stage_b(W1t, bstage[0], 0, t);     // prologue: stage k-chunk 0

    v8f acc[16];
    #pragma unroll
    for (int n = 0; n < 16; ++n)
        #pragma unroll
        for (int r = 0; r < 8; ++r) acc[n][r] = 0.f;

    for (int kc = 0; kc < 16; ++kc) {          // K chunks of 32
        if (kc + 1 < 16) {
            stage_b(W1t, bstage[(kc + 1) & 1], (kc + 1) * 32, t);
            wait_async_le<4>();                // current chunk's 4 asyncs done
        } else {
            wait_async_le<0>();
        }
        __syncthreads();

        const int kb = kc * 32;
        // A layout (16-bit 16x32): elems 0..7 -> K = kb+hf*8+j ; 8..15 -> K = kb+16+hf*8+j
        const float4 f0 = *(const float4*)(xr + kb + hf * 8);
        const float4 f1 = *(const float4*)(xr + kb + hf * 8 + 4);
        const float4 f2 = *(const float4*)(xr + kb + 16 + hf * 8);
        const float4 f3 = *(const float4*)(xr + kb + 16 + hf * 8 + 4);
        v16bf a;
        a[0]=(__bf16)f0.x; a[1]=(__bf16)f0.y; a[2]=(__bf16)f0.z; a[3]=(__bf16)f0.w;
        a[4]=(__bf16)f1.x; a[5]=(__bf16)f1.y; a[6]=(__bf16)f1.z; a[7]=(__bf16)f1.w;
        a[8]=(__bf16)f2.x; a[9]=(__bf16)f2.y; a[10]=(__bf16)f2.z; a[11]=(__bf16)f2.w;
        a[12]=(__bf16)f3.x; a[13]=(__bf16)f3.y; a[14]=(__bf16)f3.z; a[15]=(__bf16)f3.w;

        const char* bbase = &bstage[kc & 1][0];
        #pragma unroll
        for (int n = 0; n < 16; ++n) {
            // lane holds column N = n*16+m, K = kb + hf*8 + (0..7), kb + 16 + hf*8 + (0..7): 2x16B from LDS
            const char* bp = bbase + (n * 16 + m) * ROW_PAD + hf * 16;
            union { v16bf v; v8bf h[2]; } bu;
            bu.h[0] = *(const v8bf*)(bp);
            bu.h[1] = *(const v8bf*)(bp + 32);
            acc[n] = __builtin_amdgcn_wmma_f32_16x16x32_bf16(false, a, false, bu.v,
                                                             (short)0, acc[n], false, false);
        }
        __syncthreads();
    }

    // epilogue: per-lane partial s for rows M = hf*8 + r (lane covers column N = n*16+m)
    float p[8];
    #pragma unroll
    for (int r = 0; r < 8; ++r) p[r] = 0.f;
    #pragma unroll
    for (int n = 0; n < 16; ++n) {
        const float bias = b1[n * 16 + m];
        const float w2v  = W2[n * 16 + m];
        #pragma unroll
        for (int r = 0; r < 8; ++r)
            p[r] += tanhf(acc[n][r] + bias) * w2v;
    }
    #pragma unroll
    for (int r = 0; r < 8; ++r) {
        p[r] += __shfl_xor(p[r], 1, 32);
        p[r] += __shfl_xor(p[r], 2, 32);
        p[r] += __shfl_xor(p[r], 4, 32);
        p[r] += __shfl_xor(p[r], 8, 32);
    }
    if (m == 0) {
        const float b2v = b2[0];
        #pragma unroll
        for (int r = 0; r < 8; ++r)
            s_out[r0 + hf * 8 + r] = p[r] + b2v;
    }
}

// ---------------- segment softmax + weighted pooling (block per segment) ----------------
__global__ __launch_bounds__(256)
void k_pool(const float* __restrict__ sub_feat, const float* __restrict__ sub_w,
            const int* __restrict__ feat_idx, const float* __restrict__ s,
            const int* __restrict__ segs, const int* __restrict__ sege,
            __bf16* __restrict__ mol_bf, float* __restrict__ wmol_out) {
    __shared__ float red[256];
    __shared__ float lalpha[256];
    __shared__ float wsh[8];
    const int b = blockIdx.x;
    const int tid = threadIdx.x;
    const int sv = segs[b], ev = sege[b];

    // pass 1: segment max
    float lm = -3.0e38f;
    for (int e = sv + tid; e < ev; e += 256) lm = fmaxf(lm, s[e]);
    red[tid] = lm; __syncthreads();
    for (int off = 128; off > 0; off >>= 1) {
        if (tid < off) red[tid] = fmaxf(red[tid], red[tid + off]);
        __syncthreads();
    }
    const float mseg = red[0];
    __syncthreads();

    // pass 2: sum of exp
    float lz = 0.f;
    for (int e = sv + tid; e < ev; e += 256) lz += expf(s[e] - mseg);
    red[tid] = lz; __syncthreads();
    for (int off = 128; off > 0; off >>= 1) {
        if (tid < off) red[tid] += red[tid + off];
        __syncthreads();
    }
    const float zseg = red[0] + 1e-16f;
    __syncthreads();

    // pass 3: alpha-weighted pooled embedding (thread covers columns tid, tid+256)
    float acc0 = 0.f, acc1 = 0.f, wacc = 0.f;
    for (int base = sv; base < ev; base += 256) {
        const int cnt = min(256, ev - base);
        if (tid < cnt) lalpha[tid] = expf(s[base + tid] - mseg) / zseg;
        __syncthreads();
        for (int j = 0; j < cnt; ++j) {
            const int row = feat_idx[base + j];
            const float al = lalpha[j];
            const float* xr = sub_feat + (size_t)row * DD;
            if (j + 1 < cnt) {
                const int nrow = feat_idx[base + j + 1];
                __builtin_prefetch(sub_feat + (size_t)nrow * DD + tid, 0, 0);
            }
            acc0 += al * xr[tid];
            acc1 += al * xr[256 + tid];
            if (tid < KK) wacc += al * sub_w[(size_t)row * KK + tid];
        }
        __syncthreads();
    }
    mol_bf[(size_t)b * DD + tid]       = (__bf16)acc0;
    mol_bf[(size_t)b * DD + 256 + tid] = (__bf16)acc1;

    if (tid < KK) wsh[tid] = wacc;
    __syncthreads();
    if (tid < KK) {
        float ssum = 0.f;
        #pragma unroll
        for (int k = 0; k < KK; ++k) ssum += wsh[k];
        wmol_out[(size_t)b * KK + tid] = wacc / fmaxf(ssum, 1e-12f);
    }
}

// ---------------- predictor GEMM 1: hidden = relu(mol_emb @ pW1 + pb1) via WMMA ----------------
__global__ __launch_bounds__(256)
void k_pred1(const __bf16* __restrict__ mol_bf, const __bf16* __restrict__ pW1t,
             const float* __restrict__ pb1, float* __restrict__ hidden) {
    const int lane = threadIdx.x & 31;
    const int wid  = threadIdx.x >> 5;
    const int hf   = lane >> 4;
    const int m    = lane & 15;
    const int t    = blockIdx.x * 8 + wid;    // 2048 tiles = 64 (M) x 32 (N)
    const int mt   = t >> 5;
    const int nt   = t & 31;
    const int m0 = mt * 16, n0 = nt * 16;
    const __bf16* ar = mol_bf + (size_t)(m0 + m) * DD;
    const __bf16* br = pW1t   + (size_t)(n0 + m) * DD;

    v8f acc;
    #pragma unroll
    for (int r = 0; r < 8; ++r) acc[r] = 0.f;
    for (int kc = 0; kc < 16; ++kc) {
        const int kb = kc * 32;
        union { v16bf v; v8bf h[2]; } a;
        a.h[0] = *(const v8bf*)(ar + kb + hf * 16);
        a.h[1] = *(const v8bf*)(ar + kb + hf * 16 + 8);
        const v16bf bfr = *(const v16bf*)(br + kb + hf * 16);
        acc = __builtin_amdgcn_wmma_f32_16x16x32_bf16(false, a.v, false, bfr,
                                                      (short)0, acc, false, false);
    }
    const float bias = pb1[n0 + m];
    #pragma unroll
    for (int r = 0; r < 8; ++r) {
        const float v = acc[r] + bias;
        hidden[(size_t)(m0 + hf * 8 + r) * DD + n0 + m] = v > 0.f ? v : 0.f;
    }
}

// ---------------- predictor GEMM 2 (tiny): logits = hidden @ pW2 + pb2 ----------------
__global__ void k_pred2(const float* __restrict__ hidden, const float* __restrict__ pW2,
                        const float* __restrict__ pb2, float* __restrict__ logits) {
    const int i = blockIdx.x * blockDim.x + threadIdx.x;
    if (i >= BB * TT) return;
    const int b = i / TT, t = i % TT;
    const float* hr = hidden + (size_t)b * DD;
    float acc = pb2[t];
    for (int k = 0; k < DD; ++k) acc += hr[k] * pW2[(size_t)k * TT + t];
    logits[i] = acc;
}

extern "C" void kernel_launch(void* const* d_in, const int* in_sizes, int n_in,
                              void* d_out, int out_size, void* d_ws, size_t ws_size,
                              hipStream_t stream) {
    (void)in_sizes; (void)n_in; (void)out_size; (void)ws_size;
    const float* sub_feat = (const float*)d_in[0];
    const float* sub_w    = (const float*)d_in[1];
    // d_in[2] light_counts: dead code in reference (does not feed outputs) -> skipped
    const float* W1  = (const float*)d_in[3];
    const float* b1  = (const float*)d_in[4];
    const float* W2  = (const float*)d_in[5];
    const float* b2  = (const float*)d_in[6];
    // d_in[7..10] light MLP params: skipped (dead)
    const float* pW1 = (const float*)d_in[11];
    const float* pb1 = (const float*)d_in[12];
    const float* pW2 = (const float*)d_in[13];
    const float* pb2 = (const float*)d_in[14];
    const int* batch_idx = (const int*)d_in[15];
    const int* feat_idx  = (const int*)d_in[16];
    float* out = (float*)d_out;  // [B*T logits][B*K w_mol]

    char* ws = (char*)d_ws;
    size_t off = 0;
    __bf16* W1t  = (__bf16*)(ws + off); off += (size_t)HH * DD * 2;   // 256 KB
    __bf16* pW1t = (__bf16*)(ws + off); off += (size_t)DD * DD * 2;   // 512 KB
    float*  sbuf = (float*)(ws + off);  off += (size_t)EE * 4;        // 512 KB
    int* segs = (int*)(ws + off); off += (size_t)BB * 4;
    int* sege = (int*)(ws + off); off += (size_t)BB * 4;
    __bf16* mol_bf = (__bf16*)(ws + off); off += (size_t)BB * DD * 2; // 1 MB
    float* hidden = (float*)(ws + off);  off += (size_t)BB * DD * 4;  // 2 MB

    k_transpose_bf16<<<(DD * HH + 255) / 256, 256, 0, stream>>>(W1, W1t, DD, HH);
    k_transpose_bf16<<<(DD * DD + 255) / 256, 256, 0, stream>>>(pW1, pW1t, DD, DD);
    k_seg_init<<<(BB + 255) / 256, 256, 0, stream>>>(segs, sege);
    k_seg_bounds<<<(EE + 255) / 256, 256, 0, stream>>>(batch_idx, segs, sege);
    k_score<<<EE / 128, 256, 0, stream>>>(sub_feat, feat_idx, W1t, b1, W2, b2, sbuf);
    k_pool<<<BB, 256, 0, stream>>>(sub_feat, sub_w, feat_idx, sbuf, segs, sege,
                                   mol_bf, out + (size_t)BB * TT);
    k_pred1<<<(BB / 16) * (DD / 16) / 8, 256, 0, stream>>>(mol_bf, pW1t, pb1, hidden);
    k_pred2<<<(BB * TT + 255) / 256, 256, 0, stream>>>(hidden, pW2, pb2, out);
}
